// PointNetPlusEncoder_25469156065819
// MI455X (gfx1250) — matrix operand order repaired and checked by reading, not compiled, not run on hardware
//
#include <hip/hip_runtime.h>
#include <hip/hip_bf16.h>

// ---------------------------------------------------------------------------
// PointNet++ encoder for MI455X (gfx1250, wave32, WMMA).
// Matrix work: v_wmma_f32_16x16x32_f16 (f16 in, f32 accum).
// FPS staging: global_load_async_to_lds_b128 + s_wait_asynccnt (CDNA5 async path).
// ---------------------------------------------------------------------------

typedef __attribute__((ext_vector_type(16))) _Float16 v16h;
typedef __attribute__((ext_vector_type(8)))  float    v8f;

// ISA 16-bit A-matrix 16x32 layout: lanes 0-15 hold K 0..7 (VGPR0-3) and
// K 16..23 (VGPR4-7); lanes 16-31 hold K 8..15 and 24..31.
__device__ __forceinline__ int amap_k(int h, int grp) {
  return (h < 8 ? h : h + 8) + 8 * grp;
}
// B-matrix 32x16: same K order as A (lanes 0-15 K 0..7,16..23; lanes 16-31 K 8..15,24..31).
__device__ __forceinline__ int bmap_k(int h, int grp) {
  return (h < 8 ? h : h + 8) + 8 * grp;
}

// ---------------------------------------------------------------------------
// Pointwise GEMM: Y[P,COUT] = act(X[P,Cin] @ W[COUT,Cin]^T + b)
// One wave per 16-row tile; A fragment loaded once per k-step and reused
// across all COUT/16 column tiles. P must be a multiple of 128 (true for all
// call sites), so the C store is branch-free.
// ---------------------------------------------------------------------------
template <int COUT>
__global__ void k_gemm_relu_wmma(const float* __restrict__ X,
                                 const float* __restrict__ W,
                                 const float* __restrict__ bias,
                                 float* __restrict__ Y,
                                 int P, int Cin, int relu) {
  int wave = threadIdx.x >> 5;
  int m0 = (blockIdx.x * 8 + wave) * 16;
  if (m0 >= P) return;
  int lane = threadIdx.x & 31;
  int grp = lane >> 4, nn = lane & 15;
  const float* xrow = X + (long)(m0 + nn) * Cin;

  v8f acc[COUT / 16];
#pragma unroll
  for (int nt = 0; nt < COUT / 16; ++nt) acc[nt] = (v8f){};

  for (int k0 = 0; k0 < Cin; k0 += 32) {
    v16h a;
#pragma unroll
    for (int h = 0; h < 16; ++h) a[h] = (_Float16)xrow[k0 + amap_k(h, grp)];
#pragma unroll
    for (int nt = 0; nt < COUT / 16; ++nt) {
      const float* wrow = W + (long)(nt * 16 + nn) * Cin;
      v16h b;
#pragma unroll
      for (int h = 0; h < 16; ++h) b[h] = (_Float16)wrow[k0 + bmap_k(h, grp)];
      acc[nt] = __builtin_amdgcn_wmma_f32_16x16x32_f16(false, a, false, b,
                                                       (short)0, acc[nt],
                                                       false, false);
    }
  }
#pragma unroll
  for (int nt = 0; nt < COUT / 16; ++nt) {
    float bv = bias ? bias[nt * 16 + nn] : 0.0f;
#pragma unroll
    for (int e = 0; e < 8; ++e) {
      int m = m0 + e + 8 * grp;            // documented C/D layout
      float v = acc[nt][e] + bv;
      if (relu) v = fmaxf(v, 0.0f);
      Y[(long)m * COUT + nt * 16 + nn] = v;
    }
  }
}

// ---------------------------------------------------------------------------
// Input layer: [P,3] -> relu([P,32])   (K=3: scalar, memory bound)
// ---------------------------------------------------------------------------
__global__ void k_linear3_relu(const float* __restrict__ x,
                               const float* __restrict__ w,
                               const float* __restrict__ b,
                               float* __restrict__ y, long P) {
  long t = (long)blockIdx.x * blockDim.x + threadIdx.x;
  if (t >= P) return;
  float a0 = x[t * 3 + 0], a1 = x[t * 3 + 1], a2 = x[t * 3 + 2];
#pragma unroll
  for (int o = 0; o < 32; ++o) {
    y[t * 32 + o] =
        fmaxf(w[o * 3 + 0] * a0 + w[o * 3 + 1] * a1 + w[o * 3 + 2] * a2 + b[o], 0.0f);
  }
}

// ---------------------------------------------------------------------------
// Column reduction over axis 1: in [B,P,C] -> out [B,C] (mean or max)
// ---------------------------------------------------------------------------
__global__ void k_colreduce(const float* __restrict__ in, float* __restrict__ out,
                            int P, int C, int ismax) {
  __shared__ float red[256];
  int b = blockIdx.x / C, c = blockIdx.x % C;
  const float* base = in + (long)b * P * C + c;
  float acc = ismax ? -3.0e38f : 0.0f;
  for (int p = threadIdx.x; p < P; p += blockDim.x) {
    float v = base[(long)p * C];
    acc = ismax ? fmaxf(acc, v) : (acc + v);
  }
  red[threadIdx.x] = acc;
  __syncthreads();
  for (int off = blockDim.x >> 1; off > 0; off >>= 1) {
    if ((int)threadIdx.x < off) {
      float o = red[threadIdx.x + off];
      red[threadIdx.x] = ismax ? fmaxf(red[threadIdx.x], o) : (red[threadIdx.x] + o);
    }
    __syncthreads();
  }
  if (threadIdx.x == 0) out[blockIdx.x] = ismax ? red[0] : red[0] / (float)P;
}

// SE gate: gate = sigmoid(relu(mean@w1^T+b1)@w2^T+b2); one thread per (b,c)
__global__ void k_se_gate(const float* __restrict__ mean,
                          const float* __restrict__ w1, const float* __restrict__ b1,
                          const float* __restrict__ w2, const float* __restrict__ b2,
                          float* __restrict__ gate, int Bn, int C) {
  int t = blockIdx.x * blockDim.x + threadIdx.x;
  if (t >= Bn * C) return;
  int b = t / C, c = t % C;
  const float* s = mean + (long)b * C;
  float h[8];
#pragma unroll
  for (int j = 0; j < 8; ++j) {
    float acc = b1[j];
    for (int cc = 0; cc < C; ++cc) acc += s[cc] * w1[j * C + cc];
    h[j] = fmaxf(acc, 0.0f);
  }
  float g = b2[c];
#pragma unroll
  for (int j = 0; j < 8; ++j) g += h[j] * w2[c * 8 + j];
  gate[t] = 1.0f / (1.0f + __expf(-g));
}

__global__ void k_scale(float* __restrict__ f, const float* __restrict__ gate,
                        long P, int C, long total) {
  long t = (long)blockIdx.x * blockDim.x + threadIdx.x;
  if (t >= total) return;
  long b = t / (P * (long)C);
  int c = (int)(t % C);
  f[t] *= gate[b * C + c];
}

// ---------------------------------------------------------------------------
// Farthest-point sampling: one workgroup per batch. xyz is staged into LDS
// once via the CDNA5 async-to-LDS path (ASYNCcnt), then 1023 dependent
// argmax steps run entirely out of LDS (xyz + running dists + reduction:
// Np*16 + 2KB bytes of dynamic LDS; 130KB for Np=8192, within the 320KB WGP).
// ---------------------------------------------------------------------------
__global__ void k_fps(const float* __restrict__ xyz, int Np, int S,
                      int* __restrict__ idx_out, float* __restrict__ centers_out) {
  extern __shared__ float smem[];
  float* sx = smem;                 // [Np*3] staged coordinates
  float* sd = smem + (long)Np * 3;  // [Np]   running min-distances
  float* rv = sd + Np;              // [256]  argmax reduction values
  int*   ri = (int*)(rv + 256);     // [256]  argmax reduction indices

  int b = blockIdx.x;
  const float* X = xyz + (long)b * Np * 3;

  // ---- async stage xyz -> LDS (16B granules; Np*12 is 16B-divisible) ----
  {
    unsigned nb = (unsigned)Np * 12u;
    unsigned sxoff = (unsigned)(size_t)sx;   // low 32 bits of flat LDS addr = LDS offset
    for (unsigned o = (unsigned)threadIdx.x * 16u; o < nb;
         o += (unsigned)blockDim.x * 16u) {
      unsigned l = sxoff + o;
      const void* g = (const char*)X + o;
      asm volatile("global_load_async_to_lds_b128 %0, %1, off"
                   :: "v"(l), "v"(g) : "memory");
    }
    asm volatile("s_wait_asynccnt 0" ::: "memory");
  }
  for (int i = threadIdx.x; i < Np; i += blockDim.x) sd[i] = 1e10f;
  __syncthreads();

  if (threadIdx.x == 0) {
    idx_out[(long)b * S] = 0;
    centers_out[((long)b * S) * 3 + 0] = sx[0];
    centers_out[((long)b * S) * 3 + 1] = sx[1];
    centers_out[((long)b * S) * 3 + 2] = sx[2];
  }
  __syncthreads();

  int last = 0;
  for (int s = 1; s < S; ++s) {
    float px = sx[last * 3 + 0], py = sx[last * 3 + 1], pz = sx[last * 3 + 2];
    float bestv = -1.0f; int besti = 0;
    for (int i = threadIdx.x; i < Np; i += blockDim.x) {
      float dx = sx[i * 3 + 0] - px, dy = sx[i * 3 + 1] - py, dz = sx[i * 3 + 2] - pz;
      float d = dx * dx + dy * dy + dz * dz;
      float m = fminf(sd[i], d);
      sd[i] = m;
      if (m > bestv) { bestv = m; besti = i; }   // strict '>' => first max (argmax)
    }
    rv[threadIdx.x] = bestv; ri[threadIdx.x] = besti;
    __syncthreads();
    for (int off = blockDim.x >> 1; off > 0; off >>= 1) {
      if ((int)threadIdx.x < off) {
        float v2 = rv[threadIdx.x + off]; int i2 = ri[threadIdx.x + off];
        if (v2 > rv[threadIdx.x] ||
            (v2 == rv[threadIdx.x] && i2 < ri[threadIdx.x])) {
          rv[threadIdx.x] = v2; ri[threadIdx.x] = i2;
        }
      }
      __syncthreads();
    }
    last = ri[0];
    if (threadIdx.x == 0) {
      idx_out[(long)b * S + s] = last;
      centers_out[((long)b * S + s) * 3 + 0] = sx[last * 3 + 0];
      centers_out[((long)b * S + s) * 3 + 1] = sx[last * 3 + 1];
      centers_out[((long)b * S + s) * 3 + 2] = sx[last * 3 + 2];
    }
    __syncthreads();
  }
}

// ---------------------------------------------------------------------------
// Ball query: first K=32 indices within radius, pad with first hit (or 0).
// One thread per (b, center).
// ---------------------------------------------------------------------------
__global__ void k_ballquery(const float* __restrict__ xyz,
                            const float* __restrict__ centers,
                            int* __restrict__ nidx,
                            int Np, int S, int Bn, float r2) {
  int t = blockIdx.x * blockDim.x + threadIdx.x;
  if (t >= Bn * S) return;
  int b = t / S;
  const float* X = xyz + (long)b * Np * 3;
  float cx = centers[(long)t * 3 + 0];
  float cy = centers[(long)t * 3 + 1];
  float cz = centers[(long)t * 3 + 2];
  int* dst = nidx + (long)t * 32;
  int cnt = 0, first = 0;
  for (int i = 0; i < Np; ++i) {
    __builtin_prefetch(X + (long)i * 3 + 192, 0, 1);   // global_prefetch_b8
    float dx = X[i * 3 + 0] - cx, dy = X[i * 3 + 1] - cy, dz = X[i * 3 + 2] - cz;
    if (dx * dx + dy * dy + dz * dz <= r2) {
      if (cnt == 0) first = i;
      dst[cnt] = i;
      if (++cnt >= 32) break;
    }
  }
  for (int j = cnt; j < 32; ++j) dst[j] = first;
}

// ---------------------------------------------------------------------------
// Fused SA module MLP: gather neighbors -> [32 x 35] tile in LDS (f16, padded
// to K=64) -> WMMA 35->32 relu -> WMMA 32->C1 relu -> max over 32 neighbors.
// One wave per center, 8 centers per wave, 4 waves per block.
// ---------------------------------------------------------------------------
template <int C1>
__global__ void k_sa_mlp_wmma(const float* __restrict__ feat,
                              const float* __restrict__ xyz,
                              const float* __restrict__ centers,
                              const int* __restrict__ nidx,
                              const float* __restrict__ w0, const float* __restrict__ b0,
                              const float* __restrict__ w1, const float* __restrict__ b1,
                              float* __restrict__ out, int Np, int S) {
  __shared__ _Float16 s_h[4][32][64];    // grouped input tile (padded 35->64)
  __shared__ _Float16 s_h2[4][32][32];   // layer-0 activations
  int wv = threadIdx.x >> 5;
  int lane = threadIdx.x & 31;
  int grp = lane >> 4, nn = lane & 15;
  int b = blockIdx.y;

  // Hoist weight B-fragments (reused for all 8 centers handled by this wave).
  v16h bw0[2][2];
#pragma unroll
  for (int nt = 0; nt < 2; ++nt) {
    const float* row = w0 + (long)(nt * 16 + nn) * 35;
#pragma unroll
    for (int ks = 0; ks < 2; ++ks) {
      v16h f;
#pragma unroll
      for (int h = 0; h < 16; ++h) {
        int k = ks * 32 + bmap_k(h, grp);
        f[h] = (k < 35) ? (_Float16)row[k] : (_Float16)0.0f;
      }
      bw0[nt][ks] = f;
    }
  }
  v16h bw1[C1 / 16];
#pragma unroll
  for (int nt = 0; nt < C1 / 16; ++nt) {
    const float* row = w1 + (long)(nt * 16 + nn) * 32;
    v16h f;
#pragma unroll
    for (int h = 0; h < 16; ++h) f[h] = (_Float16)row[bmap_k(h, grp)];
    bw1[nt] = f;
  }

  for (int i = 0; i < 8; ++i) {
    int s = (blockIdx.x * 4 + wv) * 8 + i;
    // ---- gather: lane = neighbor ----
    int ni = nidx[((long)b * S + s) * 32 + lane];
    const float* fr = feat + ((long)b * Np + ni) * 32;
#pragma unroll
    for (int c = 0; c < 32; ++c) s_h[wv][lane][c] = (_Float16)fr[c];
    const float* cp = centers + ((long)b * S + s) * 3;
    const float* pp = xyz + ((long)b * Np + ni) * 3;
    s_h[wv][lane][32] = (_Float16)(pp[0] - cp[0]);
    s_h[wv][lane][33] = (_Float16)(pp[1] - cp[1]);
    s_h[wv][lane][34] = (_Float16)(pp[2] - cp[2]);
#pragma unroll
    for (int c = 35; c < 64; ++c) s_h[wv][lane][c] = (_Float16)0.0f;
    // Same-wave LDS ops are in-order: no barrier needed (per-wave tiles).

    // ---- layer 0: 35(->64) -> 32, relu, into s_h2 ----
#pragma unroll
    for (int mt = 0; mt < 2; ++mt) {
#pragma unroll
      for (int nt = 0; nt < 2; ++nt) {
        v8f c0 = {};
#pragma unroll
        for (int ks = 0; ks < 2; ++ks) {
          v16h a;
          int m = mt * 16 + nn;
#pragma unroll
          for (int h = 0; h < 16; ++h) a[h] = s_h[wv][m][ks * 32 + amap_k(h, grp)];
          c0 = __builtin_amdgcn_wmma_f32_16x16x32_f16(false, a, false, bw0[nt][ks],
                                                      (short)0, c0, false, false);
        }
        float bias0 = b0[nt * 16 + nn];
#pragma unroll
        for (int e = 0; e < 8; ++e) {
          int m = mt * 16 + e + 8 * grp;
          s_h2[wv][m][nt * 16 + nn] = (_Float16)fmaxf(c0[e] + bias0, 0.0f);
        }
      }
    }

    // ---- layer 1: 32 -> C1, relu, max over the 32 neighbors ----
#pragma unroll
    for (int nt = 0; nt < C1 / 16; ++nt) {
      float bias1 = b1[nt * 16 + nn];
      float pmax = -3.0e38f;
#pragma unroll
      for (int mt = 0; mt < 2; ++mt) {
        v16h a;
        int m = mt * 16 + nn;
#pragma unroll
        for (int h = 0; h < 16; ++h) a[h] = s_h2[wv][m][amap_k(h, grp)];
        v8f c1 = {};
        c1 = __builtin_amdgcn_wmma_f32_16x16x32_f16(false, a, false, bw1[nt],
                                                    (short)0, c1, false, false);
#pragma unroll
        for (int e = 0; e < 8; ++e) pmax = fmaxf(pmax, fmaxf(c1[e] + bias1, 0.0f));
      }
      pmax = fmaxf(pmax, __shfl_xor(pmax, 16, 32));   // combine row-halves
      if (lane < 16) out[((long)b * S + s) * C1 + nt * 16 + nn] = pmax;
    }
  }
}

// ---------------------------------------------------------------------------
// QKV projection (d=8,8,32) — skinny, scalar.
// ---------------------------------------------------------------------------
__global__ void k_qkv(const float* __restrict__ f,
                      const float* __restrict__ wq, const float* __restrict__ wk,
                      const float* __restrict__ wv,
                      float* __restrict__ q, float* __restrict__ k,
                      float* __restrict__ v, long P) {
  long t = (long)blockIdx.x * blockDim.x + threadIdx.x;
  if (t >= P) return;
  float fr[32];
#pragma unroll
  for (int c = 0; c < 32; ++c) fr[c] = f[t * 32 + c];
#pragma unroll
  for (int j = 0; j < 8; ++j) {
    float aq = 0.0f, ak = 0.0f;
#pragma unroll
    for (int c = 0; c < 32; ++c) { aq += fr[c] * wq[j * 32 + c]; ak += fr[c] * wk[j * 32 + c]; }
    q[t * 8 + j] = aq; k[t * 8 + j] = ak;
  }
  for (int j = 0; j < 32; ++j) {
    float a = 0.0f;
#pragma unroll
    for (int c = 0; c < 32; ++c) a += fr[c] * wv[j * 32 + c];
    v[t * 32 + j] = a;
  }
}

// ---------------------------------------------------------------------------
// Residual self-attention: one wave per query row, scores in LDS, d=8, n=1024.
// ---------------------------------------------------------------------------
__global__ void k_attention(const float* __restrict__ f, const float* __restrict__ q,
                            const float* __restrict__ k, const float* __restrict__ v,
                            const float* __restrict__ gammap,
                            float* __restrict__ out, int n) {
  __shared__ float sc[8][1024];
  int wv = threadIdx.x >> 5, lane = threadIdx.x & 31;
  int b = blockIdx.y;
  int r = blockIdx.x * 8 + wv;
  const float* qp = q + ((long)b * n + r) * 8;
  float qr[8];
#pragma unroll
  for (int c = 0; c < 8; ++c) qr[c] = qp[c];
  const float scale = 0.35355339059327373f;  // 1/sqrt(8)
  const float* kb = k + (long)b * n * 8;
  for (int j = lane; j < n; j += 32) {
    const float* kp = kb + (long)j * 8;
    float s = 0.0f;
#pragma unroll
    for (int c = 0; c < 8; ++c) s += qr[c] * kp[c];
    sc[wv][j] = s * scale;
  }
  float m = -3.0e38f;
  for (int j = lane; j < n; j += 32) m = fmaxf(m, sc[wv][j]);
#pragma unroll
  for (int o = 16; o > 0; o >>= 1) m = fmaxf(m, __shfl_xor(m, o, 32));
  float sum = 0.0f;
  for (int j = lane; j < n; j += 32) {
    float p = __expf(sc[wv][j] - m);
    sc[wv][j] = p;
    sum += p;
  }
#pragma unroll
  for (int o = 16; o > 0; o >>= 1) sum += __shfl_xor(sum, o, 32);
  const float* vb = v + (long)b * n * 32;
  float acc = 0.0f;
  for (int j = 0; j < n; ++j) {
    __builtin_prefetch(vb + (long)(j + 8) * 32, 0, 1);
    acc += sc[wv][j] * vb[(long)j * 32 + lane];   // sc broadcast, v coalesced
  }
  float g = gammap[0];
  long o = ((long)b * n + r) * 32 + lane;
  out[o] = f[o] + g * (acc / sum);
}

// Final: out[b,z] = gmax[b,:] . mlp_w[z,:] + mlp_b[z]
__global__ void k_final(const float* __restrict__ g, const float* __restrict__ w,
                        const float* __restrict__ bias, float* __restrict__ out,
                        int Bn, int Cin, int Z) {
  int t = blockIdx.x * blockDim.x + threadIdx.x;
  if (t >= Bn * Z) return;
  int b = t / Z, z = t % Z;
  float acc = bias[z];
  for (int c = 0; c < Cin; ++c) acc += g[b * Cin + c] * w[z * Cin + c];
  out[t] = acc;
}

// ---------------------------------------------------------------------------
// Host orchestration
// ---------------------------------------------------------------------------
extern "C" void kernel_launch(void* const* d_in, const int* in_sizes, int n_in,
                              void* d_out, int out_size, void* d_ws, size_t ws_size,
                              hipStream_t stream) {
  const int Bn = 8, N = 8192, S1 = 1024, S2 = 256, ZD = 256;
  (void)in_sizes; (void)n_in; (void)out_size; (void)ws_size;

  const float* x      = (const float*)d_in[0];
  const float* c1_w0  = (const float*)d_in[1];
  const float* c1_b0  = (const float*)d_in[2];
  const float* c1_w1  = (const float*)d_in[3];
  const float* c1_b1  = (const float*)d_in[4];
  const float* se1_w1 = (const float*)d_in[5];
  const float* se1_b1 = (const float*)d_in[6];
  const float* se1_w2 = (const float*)d_in[7];
  const float* se1_b2 = (const float*)d_in[8];
  const float* sa1_w0 = (const float*)d_in[9];
  const float* sa1_b0 = (const float*)d_in[10];
  const float* sa1_w1 = (const float*)d_in[11];
  const float* sa1_b1 = (const float*)d_in[12];
  const float* c2_w0  = (const float*)d_in[13];
  const float* c2_b0  = (const float*)d_in[14];
  const float* se2_w1 = (const float*)d_in[15];
  const float* se2_b1 = (const float*)d_in[16];
  const float* se2_w2 = (const float*)d_in[17];
  const float* se2_b2 = (const float*)d_in[18];
  const float* at_wq  = (const float*)d_in[19];
  const float* at_wk  = (const float*)d_in[20];
  const float* at_wv  = (const float*)d_in[21];
  const float* at_g   = (const float*)d_in[22];
  const float* sa2_w0 = (const float*)d_in[23];
  const float* sa2_b0 = (const float*)d_in[24];
  const float* sa2_w1 = (const float*)d_in[25];
  const float* sa2_b1 = (const float*)d_in[26];
  const float* mlp_w  = (const float*)d_in[27];
  const float* mlp_b  = (const float*)d_in[28];

  // FPS needs Np*16 + 2KB of dynamic LDS (130KB for Np=8192; WGP has 320KB).
  // Host-side attribute set, idempotent and capture-safe (no stream work).
  (void)hipFuncSetAttribute((const void*)k_fps,
                            hipFuncAttributeMaxDynamicSharedMemorySize,
                            N * 16 + 2048);

  // Bump allocator over workspace (~22 MB total).
  char* wp = (char*)d_ws;
  auto alloc = [&](size_t bytes) -> void* {
    void* r = (void*)wp;
    wp += (bytes + 255) & ~size_t(255);
    return r;
  };
  float* featA = (float*)alloc((size_t)Bn * N * 32 * 4);
  float* featB = (float*)alloc((size_t)Bn * N * 32 * 4);
  float* mean1 = (float*)alloc((size_t)Bn * 32 * 4);
  float* gate1 = (float*)alloc((size_t)Bn * 32 * 4);
  int*   fidx1 = (int*)  alloc((size_t)Bn * S1 * 4);
  float* cent1 = (float*)alloc((size_t)Bn * S1 * 3 * 4);
  int*   nix1  = (int*)  alloc((size_t)Bn * S1 * 32 * 4);
  float* sa1   = (float*)alloc((size_t)Bn * S1 * 32 * 4);
  float* feat2 = (float*)alloc((size_t)Bn * S1 * 32 * 4);
  float* mean2 = (float*)alloc((size_t)Bn * 32 * 4);
  float* gate2 = (float*)alloc((size_t)Bn * 32 * 4);
  float* qb    = (float*)alloc((size_t)Bn * S1 * 8 * 4);
  float* kb    = (float*)alloc((size_t)Bn * S1 * 8 * 4);
  float* vb    = (float*)alloc((size_t)Bn * S1 * 32 * 4);
  float* att   = (float*)alloc((size_t)Bn * S1 * 32 * 4);
  int*   fidx2 = (int*)  alloc((size_t)Bn * S2 * 4);
  float* cent2 = (float*)alloc((size_t)Bn * S2 * 3 * 4);
  int*   nix2  = (int*)  alloc((size_t)Bn * S2 * 32 * 4);
  float* sa2   = (float*)alloc((size_t)Bn * S2 * 64 * 4);
  float* gmax  = (float*)alloc((size_t)Bn * 64 * 4);

  long P1 = (long)Bn * N;    // 65536 points
  long P2 = (long)Bn * S1;   // 8192 points

  // ---- block 1 ----
  k_linear3_relu<<<(int)((P1 + 255) / 256), 256, 0, stream>>>(x, c1_w0, c1_b0, featA, P1);
  k_gemm_relu_wmma<32><<<(int)(P1 / 128), 256, 0, stream>>>(featA, c1_w1, c1_b1, featB,
                                                            (int)P1, 32, 1);
  k_colreduce<<<Bn * 32, 256, 0, stream>>>(featB, mean1, N, 32, 0);
  k_se_gate<<<1, 256, 0, stream>>>(mean1, se1_w1, se1_b1, se1_w2, se1_b2, gate1, Bn, 32);
  k_scale<<<(int)((P1 * 32 + 255) / 256), 256, 0, stream>>>(featB, gate1, (long)N, 32, P1 * 32);
  k_fps<<<Bn, 256, N * 16 + 2048, stream>>>(x, N, S1, fidx1, cent1);
  k_ballquery<<<(Bn * S1 + 255) / 256, 256, 0, stream>>>(x, cent1, nix1, N, S1, Bn, 0.01f);
  k_sa_mlp_wmma<32><<<dim3(S1 / 32, Bn), 128, 0, stream>>>(featB, x, cent1, nix1,
                                                           sa1_w0, sa1_b0, sa1_w1, sa1_b1,
                                                           sa1, N, S1);
  // ---- block 2 ----
  k_gemm_relu_wmma<32><<<(int)(P2 / 128), 256, 0, stream>>>(sa1, c2_w0, c2_b0, feat2,
                                                            (int)P2, 32, 1);
  k_colreduce<<<Bn * 32, 256, 0, stream>>>(feat2, mean2, S1, 32, 0);
  k_se_gate<<<1, 256, 0, stream>>>(mean2, se2_w1, se2_b1, se2_w2, se2_b2, gate2, Bn, 32);
  k_scale<<<(int)((P2 * 32 + 255) / 256), 256, 0, stream>>>(feat2, gate2, (long)S1, 32, P2 * 32);
  k_qkv<<<(int)((P2 + 255) / 256), 256, 0, stream>>>(feat2, at_wq, at_wk, at_wv, qb, kb, vb, P2);
  k_attention<<<dim3(S1 / 8, Bn), 256, 0, stream>>>(feat2, qb, kb, vb, at_g, att, S1);
  k_fps<<<Bn, 256, S1 * 16 + 2048, stream>>>(cent1, S1, S2, fidx2, cent2);
  k_ballquery<<<(Bn * S2 + 255) / 256, 256, 0, stream>>>(cent1, cent2, nix2, S1, S2, Bn, 0.04f);
  k_sa_mlp_wmma<64><<<dim3(S2 / 32, Bn), 128, 0, stream>>>(att, cent1, cent2, nix2,
                                                           sa2_w0, sa2_b0, sa2_w1, sa2_b1,
                                                           sa2, S1, S2);
  // ---- head ----
  k_colreduce<<<Bn * 64, 256, 0, stream>>>(sa2, gmax, S2, 64, 1);
  k_final<<<(Bn * ZD + 255) / 256, 256, 0, stream>>>(gmax, mlp_w, mlp_b, (float*)d_out,
                                                     Bn, 64, ZD);
}
